// FullAttention_3788161155768
// MI455X (gfx1250) — matrix operand order repaired and checked by reading, not compiled, not run on hardware
//
#include <hip/hip_runtime.h>
#include <hip/hip_bf16.h>
#include <cstddef>

typedef _Float16 v16h __attribute__((ext_vector_type(16)));
typedef float    v8f  __attribute__((ext_vector_type(8)));
typedef float    v4f  __attribute__((ext_vector_type(4)));

union H16 { v16h v; v4f f4[2]; };

#define WMMA_F16(a, b, c) \
  __builtin_amdgcn_wmma_f32_16x16x32_f16(false, (a), false, (b), (short)0, (c), false, false)

// ---------------------------------------------------------------- helpers

__device__ __forceinline__ float half_max16(float v) {
  v = fmaxf(v, __shfl_xor(v, 1));
  v = fmaxf(v, __shfl_xor(v, 2));
  v = fmaxf(v, __shfl_xor(v, 4));
  v = fmaxf(v, __shfl_xor(v, 8));
  return v;
}
__device__ __forceinline__ float half_sum16(float v) {
  v += __shfl_xor(v, 1);
  v += __shfl_xor(v, 2);
  v += __shfl_xor(v, 4);
  v += __shfl_xor(v, 8);
  return v;
}
__device__ __forceinline__ float wave_sum32(float v) {
  v += __shfl_xor(v, 1);
  v += __shfl_xor(v, 2);
  v += __shfl_xor(v, 4);
  v += __shfl_xor(v, 8);
  v += __shfl_xor(v, 16);
  return v;
}

// Issue two async 16B global->LDS copies (32B per lane, contiguous).
__device__ __forceinline__ void async_copy32(unsigned lds_addr, const _Float16* gptr) {
  unsigned long long ga = (unsigned long long)(size_t)gptr;
  asm volatile(
      "global_load_async_to_lds_b128 %0, %1, off\n\t"
      "global_load_async_to_lds_b128 %0, %1, off offset:16"
      :: "v"(lds_addr), "v"(ga) : "memory");
}
__device__ __forceinline__ void wait_async0() {
  asm volatile("s_wait_asynccnt 0x0" ::: "memory");
}

// ---------------------------------------------------------------- kernels

// Transpose fp32 [K,N] weight -> f16 [N,K]
__global__ void tconv_kernel(const float* __restrict__ W, _Float16* __restrict__ Wt,
                             int K, int N) {
  int idx = blockIdx.x * 256 + threadIdx.x;
  if (idx >= K * N) return;
  int k = idx / N, n = idx % N;
  Wt[(size_t)n * K + k] = (_Float16)W[idx];
}

// GroupNorm over (64 ch x 2048 spatial) per (b, group); writes xn16 [B*S, C] f16.
__global__ void gn_kernel(const float* __restrict__ x, const float* __restrict__ w,
                          const float* __restrict__ bia, _Float16* __restrict__ xn) {
  __shared__ float sh1[256];
  __shared__ float sh2[256];
  const int tid = threadIdx.x;
  const int b = blockIdx.x / 12, g = blockIdx.x % 12;
  const float* xb = x + ((size_t)(b * 768 + g * 64)) * 2048;
  float sum = 0.f, ssq = 0.f;
  for (int i = tid; i < 64 * 2048; i += 256) {
    float v = xb[i];
    sum += v;
    ssq += v * v;
  }
  sh1[tid] = sum; sh2[tid] = ssq;
  __syncthreads();
  for (int o = 128; o > 0; o >>= 1) {
    if (tid < o) { sh1[tid] += sh1[tid + o]; sh2[tid] += sh2[tid + o]; }
    __syncthreads();
  }
  const float inv_n = 1.0f / (64.0f * 2048.0f);
  const float mu = sh1[0] * inv_n;
  const float var = sh2[0] * inv_n - mu * mu;
  const float rstd = rsqrtf(var + 1e-5f);
  for (int i = tid; i < 64 * 2048; i += 256) {
    int cc = i >> 11, s = i & 2047;
    int c = g * 64 + cc;
    float v = xb[i];
    xn[((size_t)(b * 2048 + s)) * 768 + c] = (_Float16)((v - mu) * rstd * w[c] + bia[c]);
  }
}

// Tiled WMMA GEMM, async-to-LDS double-buffered software pipeline.
// acc += A1[M,K1] * B1t[N,K1]^T (+ A2[M,K2] * B2t[N,K2]^T if A2 != null; the
// flattened iteration space pipelines across the pass boundary).
// mode 0: C16[row*N+col] = f16(acc + bias[col])
// mode 1: out32[(b*768+col)*2048+s] = (acc + bias[col])*gamma[col] + xin[...]
__global__ __launch_bounds__(256)
void gemm_wmma_kernel(const _Float16* __restrict__ A1, const _Float16* __restrict__ B1,
                      int K1,
                      const _Float16* __restrict__ A2, const _Float16* __restrict__ B2,
                      int K2,
                      int N, const float* __restrict__ bias, int mode,
                      _Float16* __restrict__ C16, const float* __restrict__ gamma,
                      const float* __restrict__ xin, float* __restrict__ out32) {
  __shared__ __align__(16) _Float16 At[2][128 * 40];
  __shared__ __align__(16) _Float16 Bt[2][128 * 40];
  const int tid = threadIdx.x, lane = tid & 31, wid = tid >> 5;
  const int waveM = (wid & 1) * 64, waveN = (wid >> 1) * 32;
  const int tile_m = blockIdx.y * 128, tile_n = blockIdx.x * 128;
  const int rlane = lane & 15, khalf = lane >> 4;
  const int lrow = tid >> 1, lk = (tid & 1) << 4;

  const int iters1 = K1 >> 5;
  const int iters2 = (A2 != nullptr) ? (K2 >> 5) : 0;
  const int total = iters1 + iters2;

  v8f acc[4][2];
#pragma unroll
  for (int i = 0; i < 4; ++i)
#pragma unroll
    for (int j = 0; j < 2; ++j)
#pragma unroll
      for (int r = 0; r < 8; ++r) acc[i][j][r] = 0.f;

  auto issue_tile = [&](int it, int buf) {
    const _Float16 *ga, *gb;
    if (it < iters1) {
      ga = A1 + (size_t)(tile_m + lrow) * K1 + (it << 5) + lk;
      gb = B1 + (size_t)(tile_n + lrow) * K1 + (it << 5) + lk;
    } else {
      int j = it - iters1;
      ga = A2 + (size_t)(tile_m + lrow) * K2 + (j << 5) + lk;
      gb = B2 + (size_t)(tile_n + lrow) * K2 + (j << 5) + lk;
    }
    unsigned la = (unsigned)(size_t)&At[buf][lrow * 40 + lk];
    unsigned lb = (unsigned)(size_t)&Bt[buf][lrow * 40 + lk];
    async_copy32(la, ga);   // A tile slice: straight to LDS, no VGPR staging
    async_copy32(lb, gb);   // B tile slice
  };

  issue_tile(0, 0);
  for (int it = 0; it < total; ++it) {
    const int buf = it & 1;
    wait_async0();        // this wave's slices of tile `it` are in LDS
    __syncthreads();      // ... and everyone else's too
    if (it + 1 < total) issue_tile(it + 1, buf ^ 1);  // overlap with compute

    H16 af[4], bf[2];
#pragma unroll
    for (int mt = 0; mt < 4; ++mt) {
      const _Float16* p = &At[buf][(waveM + mt * 16 + rlane) * 40 + khalf * 8];
      af[mt].f4[0] = *(const v4f*)p;
      af[mt].f4[1] = *(const v4f*)(p + 16);
    }
#pragma unroll
    for (int nt = 0; nt < 2; ++nt) {
      const _Float16* p = &Bt[buf][(waveN + nt * 16 + rlane) * 40 + khalf * 8];
      bf[nt].f4[0] = *(const v4f*)p;
      bf[nt].f4[1] = *(const v4f*)(p + 16);
    }
#pragma unroll
    for (int mt = 0; mt < 4; ++mt)
#pragma unroll
      for (int nt = 0; nt < 2; ++nt)
        acc[mt][nt] = WMMA_F16(af[mt].v, bf[nt].v, acc[mt][nt]);
  }

  // Epilogue: mode branch hoisted out of the loops; everything fully unrolled
  // so `acc` stays in VGPRs (no dynamic indexing -> no scratch spill).
  if (mode == 0) {
#pragma unroll
    for (int mt = 0; mt < 4; ++mt) {
#pragma unroll
      for (int nt = 0; nt < 2; ++nt) {
#pragma unroll
        for (int r = 0; r < 8; ++r) {
          int row = tile_m + waveM + mt * 16 + khalf * 8 + r;
          int col = tile_n + waveN + nt * 16 + rlane;
          float v = acc[mt][nt][r] + bias[col];
          C16[(size_t)row * N + col] = (_Float16)v;
        }
      }
    }
  } else {
#pragma unroll
    for (int mt = 0; mt < 4; ++mt) {
#pragma unroll
      for (int nt = 0; nt < 2; ++nt) {
#pragma unroll
        for (int r = 0; r < 8; ++r) {
          int row = tile_m + waveM + mt * 16 + khalf * 8 + r;
          int col = tile_n + waveN + nt * 16 + rlane;
          float v = (acc[mt][nt][r] + bias[col]) * gamma[col];
          int b = row >> 11, s = row & 2047;
          size_t xi = ((size_t)(b * 768 + col)) * 2048 + s;
          out32[xi] = v + xin[xi];
        }
      }
    }
  }
}

// Per (b,s,h) wave: q/k layernorm (hd=64) + axial RoPE (rot=48); v transposed copy.
__global__ __launch_bounds__(256)
void qkv_kernel(const _Float16* __restrict__ fus,
                const float* __restrict__ qw, const float* __restrict__ qb,
                const float* __restrict__ kw, const float* __restrict__ kb,
                _Float16* __restrict__ q16, _Float16* __restrict__ k16,
                _Float16* __restrict__ vt16) {
  const int lane = threadIdx.x & 31, wid = threadIdx.x >> 5;
  const int idx = blockIdx.x * 8 + wid;                 // [0, 2*2048*12)
  const int h = idx % 12, s = (idx / 12) & 2047, b = idx / (12 * 2048);
  const _Float16* base = fus + (size_t)(b * 2048 + s) * 5376;
  const int d0 = lane * 2, d1 = d0 + 1;

  float q0 = (float)base[3072 + h * 64 + d0], q1 = (float)base[3072 + h * 64 + d1];
  float k0 = (float)base[3840 + h * 64 + d0], k1 = (float)base[3840 + h * 64 + d1];
  float v0 = (float)base[4608 + h * 64 + d0], v1 = (float)base[4608 + h * 64 + d1];

  // layernorm over the 64 head dims (each lane owns 2)
  float muq = wave_sum32(q0 + q1) * (1.f / 64.f);
  float vaq = wave_sum32((q0 - muq) * (q0 - muq) + (q1 - muq) * (q1 - muq)) * (1.f / 64.f);
  float rq = rsqrtf(vaq + 1e-5f);
  float qn0 = (q0 - muq) * rq * qw[d0] + qb[d0];
  float qn1 = (q1 - muq) * rq * qw[d1] + qb[d1];

  float muk = wave_sum32(k0 + k1) * (1.f / 64.f);
  float vak = wave_sum32((k0 - muk) * (k0 - muk) + (k1 - muk) * (k1 - muk)) * (1.f / 64.f);
  float rk = rsqrtf(vak + 1e-5f);
  float kn0 = (k0 - muk) * rk * kw[d0] + kb[d0];
  float kn1 = (k1 - muk) * rk * kw[d1] + kb[d1];

  if (d0 < 48) {  // axial RoPE: pair (2l, 2l+1) shares one frequency
    int axis = d0 >> 4;
    int fi = (d0 & 15) >> 1;
    float basef = (1.0f + 127.0f * fi * (1.0f / 7.0f)) * 3.14159265358979f;
    int di = s & 7, wi = (s >> 3) & 15, hi = s >> 7;
    float pos = (axis == 0) ? hi * (2.f / 15.f) - 1.f
              : (axis == 1) ? wi * (2.f / 15.f) - 1.f
                            : di * (2.f / 7.f) - 1.f;
    float f = pos * basef;
    float cf = __cosf(f), sf = __sinf(f);
    float t0 = qn0 * cf - qn1 * sf, t1 = qn1 * cf + qn0 * sf;
    qn0 = t0; qn1 = t1;
    t0 = kn0 * cf - kn1 * sf; t1 = kn1 * cf + kn0 * sf;
    kn0 = t0; kn1 = t1;
  }

  const int bh = b * 12 + h;
  _Float16* qo = q16 + ((size_t)bh * 2048 + s) * 64;
  _Float16* ko = k16 + ((size_t)bh * 2048 + s) * 64;
  qo[d0] = (_Float16)qn0; qo[d1] = (_Float16)qn1;
  ko[d0] = (_Float16)kn0; ko[d1] = (_Float16)kn1;
  vt16[((size_t)bh * 64 + d0) * 2048 + s] = (_Float16)v0;
  vt16[((size_t)bh * 64 + d1) * 2048 + s] = (_Float16)v1;
}

// silu(gate) * xh -> g16 [4096, 1536] f16
__global__ void silu_kernel(const _Float16* __restrict__ fus, _Float16* __restrict__ g16) {
  size_t idx = (size_t)blockIdx.x * 256 + threadIdx.x;
  if (idx >= (size_t)4096 * 1536) return;
  size_t row = idx / 1536;
  int j = (int)(idx % 1536);
  float xh = (float)fus[row * 5376 + j];
  float gt = (float)fus[row * 5376 + 1536 + j];
  float sg = gt / (1.f + __expf(-gt));
  g16[idx] = (_Float16)(sg * xh);
}

// Flash attention: 1 wave = 16 q rows; kv tiles of 32; online softmax.
// K/V fragments stream from global (whole K/V set fits in the 192MB L2).
__global__ __launch_bounds__(128)
void attn_kernel(const _Float16* __restrict__ q16, const _Float16* __restrict__ k16,
                 const _Float16* __restrict__ vt16, _Float16* __restrict__ att16) {
  __shared__ __align__(16) _Float16 P[4][16 * 40];   // per-wave P scratch (C->A layout)
  const int lane = threadIdx.x & 31, wid = threadIdx.x >> 5;
  const int bh = blockIdx.x, b = bh / 12, h = bh % 12;
  const int qbase = blockIdx.y * 64 + wid * 16;
  const int rlane = lane & 15, khalf = lane >> 4;

  const _Float16* qp = q16 + (size_t)bh * 2048 * 64;
  const _Float16* kp = k16 + (size_t)bh * 2048 * 64;
  const _Float16* vp = vt16 + (size_t)bh * 64 * 2048;

  H16 qa[2];
#pragma unroll
  for (int kc = 0; kc < 2; ++kc) {
    const _Float16* p = qp + (size_t)(qbase + rlane) * 64 + kc * 32 + khalf * 8;
    qa[kc].f4[0] = *(const v4f*)p;
    qa[kc].f4[1] = *(const v4f*)(p + 16);
  }

  v8f o[4];
  float mrow[8], lrow[8];
#pragma unroll
  for (int t = 0; t < 4; ++t)
#pragma unroll
    for (int r = 0; r < 8; ++r) o[t][r] = 0.f;
#pragma unroll
  for (int r = 0; r < 8; ++r) { mrow[r] = -3.0e38f; lrow[r] = 0.f; }

  _Float16* Pw = P[wid];

  for (int kv = 0; kv < 2048; kv += 32) {
    v8f s0, s1;
#pragma unroll
    for (int r = 0; r < 8; ++r) { s0[r] = 0.f; s1[r] = 0.f; }
#pragma unroll
    for (int kc = 0; kc < 2; ++kc) {
      H16 kb0, kb1;
      const _Float16* p0 = kp + (size_t)(kv + rlane) * 64 + kc * 32 + khalf * 8;
      const _Float16* p1 = kp + (size_t)(kv + 16 + rlane) * 64 + kc * 32 + khalf * 8;
      kb0.f4[0] = *(const v4f*)p0; kb0.f4[1] = *(const v4f*)(p0 + 16);
      kb1.f4[0] = *(const v4f*)p1; kb1.f4[1] = *(const v4f*)(p1 + 16);
      s0 = WMMA_F16(qa[kc].v, kb0.v, s0);
      s1 = WMMA_F16(qa[kc].v, kb1.v, s1);
    }
#pragma unroll
    for (int r = 0; r < 8; ++r) {
      float f0 = s0[r] * 0.125f, f1 = s1[r] * 0.125f;    // 1/sqrt(64)
      float mx = half_max16(fmaxf(f0, f1));
      float mn = fmaxf(mrow[r], mx);
      float corr = __expf(mrow[r] - mn);
      mrow[r] = mn;
      f0 = __expf(f0 - mn);
      f1 = __expf(f1 - mn);
      float rs = half_sum16(f0 + f1);
      lrow[r] = lrow[r] * corr + rs;
#pragma unroll
      for (int t = 0; t < 4; ++t) o[t][r] *= corr;
      int prow = khalf * 8 + r;
      Pw[prow * 40 + rlane] = (_Float16)f0;
      Pw[prow * 40 + 16 + rlane] = (_Float16)f1;
    }
    H16 pa;
    const _Float16* pp = &Pw[rlane * 40 + khalf * 8];
    pa.f4[0] = *(const v4f*)pp;
    pa.f4[1] = *(const v4f*)(pp + 16);
#pragma unroll
    for (int t = 0; t < 4; ++t) {
      H16 vb;
      const _Float16* pv = vp + (size_t)(t * 16 + rlane) * 2048 + kv + khalf * 8;
      vb.f4[0] = *(const v4f*)pv;
      vb.f4[1] = *(const v4f*)(pv + 16);
      o[t] = WMMA_F16(pa.v, vb.v, o[t]);
    }
  }

#pragma unroll
  for (int r = 0; r < 8; ++r) {
    float inv = 1.0f / lrow[r];
    int srow = qbase + khalf * 8 + r;
    size_t orow = (size_t)(b * 2048 + srow) * 768 + h * 64;
#pragma unroll
    for (int t = 0; t < 4; ++t)
      att16[orow + t * 16 + rlane] = (_Float16)(o[t][r] * inv);
  }
}

// ---------------------------------------------------------------- launch

extern "C" void kernel_launch(void* const* d_in, const int* in_sizes, int n_in,
                              void* d_out, int out_size, void* d_ws, size_t ws_size,
                              hipStream_t stream) {
  (void)in_sizes; (void)n_in; (void)out_size; (void)ws_size;
  const float* x     = (const float*)d_in[0];
  const float* n1w   = (const float*)d_in[1];
  const float* n1b   = (const float*)d_in[2];
  const float* gamma = (const float*)d_in[3];
  const float* Wf    = (const float*)d_in[4];
  const float* bfu   = (const float*)d_in[5];
  const float* qw    = (const float*)d_in[6];
  const float* qb    = (const float*)d_in[7];
  const float* kw    = (const float*)d_in[8];
  const float* kb    = (const float*)d_in[9];
  const float* Wa    = (const float*)d_in[10];
  const float* Wfo   = (const float*)d_in[11];
  const float* bfo   = (const float*)d_in[12];
  float* out = (float*)d_out;

  char* ws = (char*)d_ws;
  size_t off = 0;
  auto take = [&](size_t elems) -> _Float16* {
    _Float16* p = (_Float16*)(ws + off);
    off += ((elems * 2 + 255) & ~(size_t)255);
    return p;
  };
  _Float16* wfT   = take((size_t)5376 * 768);   // W_fused^T  f16
  _Float16* waT   = take((size_t)768 * 768);    // W_attn^T   f16
  _Float16* wfoT  = take((size_t)768 * 1536);   // W_ffout^T  f16
  _Float16* xn16  = take((size_t)4096 * 768);   // groupnormed activations
  _Float16* fus   = take((size_t)4096 * 5376);  // fused GEMM output
  _Float16* q16   = take((size_t)24 * 2048 * 64);
  _Float16* k16   = take((size_t)24 * 2048 * 64);
  _Float16* vt16  = take((size_t)24 * 64 * 2048);
  _Float16* att16 = take((size_t)4096 * 768);
  _Float16* g16   = take((size_t)4096 * 1536);

  tconv_kernel<<<(768 * 5376 + 255) / 256, 256, 0, stream>>>(Wf, wfT, 768, 5376);
  tconv_kernel<<<(768 * 768 + 255) / 256, 256, 0, stream>>>(Wa, waT, 768, 768);
  tconv_kernel<<<(1536 * 768 + 255) / 256, 256, 0, stream>>>(Wfo, wfoT, 1536, 768);

  gn_kernel<<<24, 256, 0, stream>>>(x, n1w, n1b, xn16);

  // fused = xn @ W_fused + b_fused   (4096 x 5376, K=768)
  gemm_wmma_kernel<<<dim3(42, 32), 256, 0, stream>>>(
      xn16, wfT, 768, nullptr, nullptr, 0, 5376, bfu, 0,
      fus, nullptr, nullptr, nullptr);

  qkv_kernel<<<(2 * 2048 * 12) / 8, 256, 0, stream>>>(fus, qw, qb, kw, kb,
                                                      q16, k16, vt16);

  attn_kernel<<<dim3(24, 32), 128, 0, stream>>>(q16, k16, vt16, att16);

  silu_kernel<<<(4096 * 1536) / 256, 256, 0, stream>>>(fus, g16);

  // out = gamma*(att@W_attn + g@W_ffout + b_ffout) + x   (4096 x 768; dual-K)
  gemm_wmma_kernel<<<dim3(6, 32), 256, 0, stream>>>(
      att16, waT, 768, g16, wfoT, 1536, 768, bfo, 1,
      nullptr, gamma, x, out);
}
